// Combinedlayer_80616536146600
// MI455X (gfx1250) — matrix operand order repaired and checked by reading, not compiled, not run on hardware
//
#include <hip/hip_runtime.h>
#include <hip/hip_bf16.h>

// ---------------------------------------------------------------------------
// Problem constants (match reference)
// ---------------------------------------------------------------------------
#define NNODE 2048
#define NREL  4
#define DIN   512
#define DOUT  512
#define NHEAD 8
#define DHEAD 64
#define NEDGE 262144
#define NSEG  8192            // NNODE * NREL
#define EPSF  1e-10f
#define INV_SCALE_TEMP 0.25f  // 1 / (sqrt(64) * 0.5)

typedef __attribute__((ext_vector_type(16))) _Float16 v16h;
typedef __attribute__((ext_vector_type(8)))  float    v8f;

// ---------------------------------------------------------------------------
// WMMA helpers (contiguous k order chosen by this file, wave32)
//   A 16x32 f16: lane<16 holds row (lane), K = k..k+15
//                lane>=16 holds row (lane-16), K = k+16..k+31
//   B 32x16 f16 (B[k][j] = Wrow_j[k]): lane holds col j=lane&15,
//                K = k + (lane>=16?16:0) .. +15  (contiguous 32 bytes)
// ---------------------------------------------------------------------------
__device__ __forceinline__ v16h ld_a(const _Float16* p, int off0) {
  v16h a;
  *reinterpret_cast<int4*>(&a)       = *reinterpret_cast<const int4*>(p + off0);
  *(reinterpret_cast<int4*>(&a) + 1) = *reinterpret_cast<const int4*>(p + off0 + 8);
  return a;
}
__device__ __forceinline__ v16h ld_b(const _Float16* p, int off) {
  return *reinterpret_cast<const v16h*>(p + off);
}
__device__ __forceinline__ v8f wmma_f16(v16h a, v16h b, v8f c) {
  return __builtin_amdgcn_wmma_f32_16x16x32_f16(false, a, false, b, (short)0, c,
                                                false, false);
}

// ---------------------------------------------------------------------------
// 1) Edge histogram: deg (weighted) + count per segment
// ---------------------------------------------------------------------------
__global__ void k_edge_hist(const int* __restrict__ el,
                            const float* __restrict__ ew,
                            float* __restrict__ deg,
                            unsigned* __restrict__ cnt) {
  int e = blockIdx.x * 256 + threadIdx.x;
  if (e >= NEDGE) return;
  int s = el[e * 3 + 1] * NREL + el[e * 3 + 2];
  atomicAdd(&deg[s], ew[e]);
  atomicAdd(&cnt[s], 1u);
}

// ---------------------------------------------------------------------------
// 2) Exclusive scan of counts (8192 values, single 1024-thread block)
// ---------------------------------------------------------------------------
__global__ __launch_bounds__(1024) void k_scan(const unsigned* __restrict__ cnt,
                                               unsigned* __restrict__ offs,
                                               unsigned* __restrict__ cursor) {
  __shared__ unsigned s[1024];
  int t = threadIdx.x;
  unsigned loc[8], tot = 0;
#pragma unroll
  for (int j = 0; j < 8; ++j) { loc[j] = cnt[t * 8 + j]; tot += loc[j]; }
  s[t] = tot;
  __syncthreads();
  for (int off = 1; off < 1024; off <<= 1) {
    unsigned v = (t >= off) ? s[t - off] : 0u;
    __syncthreads();
    s[t] += v;
    __syncthreads();
  }
  unsigned run = s[t] - tot;
#pragma unroll
  for (int j = 0; j < 8; ++j) {
    offs[t * 8 + j] = run;
    cursor[t * 8 + j] = run;
    run += loc[j];
  }
}

// ---------------------------------------------------------------------------
// 3) Scatter edge ids into CSR order
// ---------------------------------------------------------------------------
__global__ void k_scatter(const int* __restrict__ el,
                          unsigned* __restrict__ cursor,
                          unsigned* __restrict__ perm) {
  int e = blockIdx.x * 256 + threadIdx.x;
  if (e >= NEDGE) return;
  int s = el[e * 3 + 1] * NREL + el[e * 3 + 2];
  unsigned p = atomicAdd(&cursor[s], 1u);
  perm[p] = (unsigned)e;
}

// ---------------------------------------------------------------------------
// 4) Segment gather-sum: update[s] = sum_e (w_e/deg) * nf[node_in[e]], -> f16
//    One block (128 threads) per segment; each thread owns 4 columns.
// ---------------------------------------------------------------------------
__global__ __launch_bounds__(128) void k_gather(
    const float* __restrict__ nf, const int* __restrict__ el,
    const float* __restrict__ ewt, const float* __restrict__ deg,
    const unsigned* __restrict__ offs, const unsigned* __restrict__ cnt,
    const unsigned* __restrict__ perm, _Float16* __restrict__ upd) {
  int s = blockIdx.x;
  int t = threadIdx.x;
  unsigned start = offs[s];
  unsigned num = cnt[s];
  float dsum = deg[s] + EPSF;
  float4 acc = {0.f, 0.f, 0.f, 0.f};
  for (unsigned i = 0; i < num; ++i) {
    unsigned e = perm[start + i];
    int ni = el[(size_t)e * 3];
    float w = ewt[e] / dsum;
    if (i + 1 < num) {
      unsigned e2 = perm[start + i + 1];
      __builtin_prefetch(nf + (size_t)el[(size_t)e2 * 3] * DIN + t * 4);
    }
    float4 v = reinterpret_cast<const float4*>(nf + (size_t)ni * DIN)[t];
    acc.x += w * v.x; acc.y += w * v.y; acc.z += w * v.z; acc.w += w * v.w;
  }
  _Float16* o = upd + (size_t)s * DIN + t * 4;
  o[0] = (_Float16)acc.x; o[1] = (_Float16)acc.y;
  o[2] = (_Float16)acc.z; o[3] = (_Float16)acc.w;
}

// ---------------------------------------------------------------------------
// 5) f32 -> f16 conversion (weights)
// ---------------------------------------------------------------------------
__global__ void k_cvt(const float* __restrict__ src, _Float16* __restrict__ dst,
                      int n) {
  int i = blockIdx.x * 256 + threadIdx.x;
  if (i < n) dst[i] = (_Float16)src[i];
}

// ---------------------------------------------------------------------------
// 6) GEMM  C[m,j] = act( sum_k A[m,k]*W[j,k] + bias[j] ), M=8192, N=K=512
//    One wave per 16x16 tile, fully unrolled K loop (16 WMMAs).
// ---------------------------------------------------------------------------
__global__ __launch_bounds__(128) void k_gemm_h(
    const _Float16* __restrict__ A, const _Float16* __restrict__ W,
    const float* __restrict__ bias, _Float16* __restrict__ C, int do_relu) {
  int wid = blockIdx.x * 4 + (threadIdx.x >> 5);
  int lane = threadIdx.x & 31, l15 = lane & 15, hi = lane >> 4;
  int m0 = (wid >> 5) << 4;  // 32 tiles along N
  int n0 = (wid & 31) << 4;
  const _Float16* arow = A + (size_t)(m0 + l15) * DIN;
  const _Float16* brow = W + (size_t)(n0 + l15) * DIN;
  int offa = hi ? 16 : 0, offb = hi ? 16 : 0;
  v8f c = {};
#pragma unroll
  for (int k = 0; k < 512; k += 32) {
    v16h a = ld_a(arow, k + offa);
    v16h b = ld_b(brow, k + offb);
    c = wmma_f16(a, b, c);
  }
  float bb = bias[n0 + l15];
#pragma unroll
  for (int v = 0; v < 8; ++v) {
    float x = c[v] + bb;
    if (do_relu) x = fmaxf(x, 0.f);
    int row = m0 + v + hi * 8;
    C[(size_t)row * DOUT + n0 + l15] = (_Float16)x;
  }
}

// ---------------------------------------------------------------------------
// 7) Attention pass 1: per (r,n,h) running max / sum-exp over all m (online).
//    One wave per (r,h,16-row strip). Scores via WMMA (K=64 -> 2 steps).
// ---------------------------------------------------------------------------
__global__ __launch_bounds__(128) void k_attn_stats(
    const _Float16* __restrict__ Q, const _Float16* __restrict__ Kmat,
    float* __restrict__ stats) {
  int gw = blockIdx.x * 4 + (threadIdx.x >> 5);
  int lane = threadIdx.x & 31, l15 = lane & 15, hi = lane >> 4;
  int ns = gw & 127, h = (gw >> 7) & 7, r = gw >> 10;
  int offa = hi ? 16 : 0, offb = hi ? 16 : 0;
  const _Float16* qrow = Q + (size_t)(r * NNODE + ns * 16 + l15) * DOUT + h * DHEAD;
  v16h a0 = ld_a(qrow, offa);
  v16h a1 = ld_a(qrow, 32 + offa);
  float mrun[8], lrun[8];
#pragma unroll
  for (int v = 0; v < 8; ++v) { mrun[v] = -1e30f; lrun[v] = 0.f; }
  for (int mt = 0; mt < 128; ++mt) {
    const _Float16* krow =
        Kmat + (size_t)(r * NNODE + mt * 16 + l15) * DOUT + h * DHEAD;
    v16h b0 = ld_b(krow, offb);
    v16h b1 = ld_b(krow, 32 + offb);
    v8f c = {};
    c = wmma_f16(a0, b0, c);
    c = wmma_f16(a1, b1, c);
#pragma unroll
    for (int v = 0; v < 8; ++v) {
      float sc = c[v] * INV_SCALE_TEMP;
      float t = sc;
      t = fmaxf(t, __shfl_xor(t, 8, 16));
      t = fmaxf(t, __shfl_xor(t, 4, 16));
      t = fmaxf(t, __shfl_xor(t, 2, 16));
      t = fmaxf(t, __shfl_xor(t, 1, 16));
      float nm = fmaxf(mrun[v], t);
      float e = __expf(sc - nm);
      e += __shfl_xor(e, 8, 16);
      e += __shfl_xor(e, 4, 16);
      e += __shfl_xor(e, 2, 16);
      e += __shfl_xor(e, 1, 16);
      lrun[v] = lrun[v] * __expf(mrun[v] - nm) + e;
      mrun[v] = nm;
    }
  }
  if (l15 == 0) {
#pragma unroll
    for (int v = 0; v < 8; ++v) {
      int n = ns * 16 + v + hi * 8;
      size_t idx = ((size_t)(r * NNODE + n) * NHEAD + h) * 2;
      stats[idx] = mrun[v];
      stats[idx + 1] = lrun[v];
    }
  }
}

// ---------------------------------------------------------------------------
// 8) Attention pass 2: A[r,n,m] = (1/8) sum_h exp(s-m_i)/l_i (stats via LDS).
//    One wave per 16x16 A tile; 8 waves/block share one (r, n-strip).
// ---------------------------------------------------------------------------
__global__ __launch_bounds__(256) void k_attn_amean(
    const _Float16* __restrict__ Q, const _Float16* __restrict__ Kmat,
    const float* __restrict__ stats, float* __restrict__ Am) {
  __shared__ float ss[256];  // 16 rows x (8h x {m,l})
  int wave = threadIdx.x >> 5;
  int gw = blockIdx.x * 8 + wave;
  int mt = gw & 127, ns = (gw >> 7) & 127, r = gw >> 14;
  {
    int rowL = threadIdx.x >> 4, rest = threadIdx.x & 15;
    ss[threadIdx.x] =
        stats[(size_t)(r * NNODE + ns * 16 + rowL) * 16 + rest];
  }
  __syncthreads();
  int lane = threadIdx.x & 31, l15 = lane & 15, hi = lane >> 4;
  int offa = hi ? 16 : 0, offb = hi ? 16 : 0;
  const _Float16* qbase = Q + (size_t)(r * NNODE + ns * 16 + l15) * DOUT;
  const _Float16* kbase = Kmat + (size_t)(r * NNODE + mt * 16 + l15) * DOUT;
  float acc[8] = {0.f, 0.f, 0.f, 0.f, 0.f, 0.f, 0.f, 0.f};
  for (int h = 0; h < NHEAD; ++h) {
    v16h a0 = ld_a(qbase + h * DHEAD, offa);
    v16h a1 = ld_a(qbase + h * DHEAD, 32 + offa);
    v16h b0 = ld_b(kbase + h * DHEAD, offb);
    v16h b1 = ld_b(kbase + h * DHEAD, 32 + offb);
    v8f c = {};
    c = wmma_f16(a0, b0, c);
    c = wmma_f16(a1, b1, c);
#pragma unroll
    for (int v = 0; v < 8; ++v) {
      int rowL = v + hi * 8;
      float mi = ss[rowL * 16 + h * 2];
      float li = ss[rowL * 16 + h * 2 + 1];
      acc[v] += __expf(c[v] * INV_SCALE_TEMP - mi) / li;
    }
  }
#pragma unroll
  for (int v = 0; v < 8; ++v) {
    int rowL = v + hi * 8;
    size_t row = (size_t)(r * NNODE + ns * 16 + rowL);
    Am[(row << 11) + mt * 16 + l15] = acc[v] * 0.125f;
  }
}

// ---------------------------------------------------------------------------
// 9) Top-3 selection per A row (stable desc by value, asc by index),
//    with the reference's 5th-value tie fallback (closest-to-i among top-5).
//    One wave per row.
// ---------------------------------------------------------------------------
__global__ __launch_bounds__(128) void k_select(const float* __restrict__ Am,
                                                int* __restrict__ selid,
                                                float* __restrict__ selval) {
  __shared__ float sv[4 * 160];
  __shared__ int si[4 * 160];
  __shared__ float sfifth[4];
  int wave = threadIdx.x >> 5, lane = threadIdx.x & 31;
  int row = blockIdx.x * 4 + wave;  // 0..8191
  int n = row & (NNODE - 1);
  const float* arow = Am + ((size_t)row << 11);

  float tv[5]; int ti[5];
#pragma unroll
  for (int j = 0; j < 5; ++j) { tv[j] = -1e30f; ti[j] = 1 << 30; }
  for (int m = lane; m < NNODE; m += 32) {
    float x = arow[m];
    if (x > tv[4] || (x == tv[4] && m < ti[4])) {
      int j = 4;
      while (j > 0 && (x > tv[j - 1] || (x == tv[j - 1] && m < ti[j - 1]))) {
        tv[j] = tv[j - 1]; ti[j] = ti[j - 1]; --j;
      }
      tv[j] = x; ti[j] = m;
    }
  }
  int base = wave * 160 + lane * 5;
#pragma unroll
  for (int j = 0; j < 5; ++j) { sv[base + j] = tv[j]; si[base + j] = ti[j]; }
  __syncthreads();

  float t5v[5]; int t5i[5];
#pragma unroll
  for (int j = 0; j < 5; ++j) { t5v[j] = -1e30f; t5i[j] = 1 << 30; }
  if (lane == 0) {
    for (int t = 0; t < 160; ++t) {
      float x = sv[wave * 160 + t]; int mi = si[wave * 160 + t];
      if (x > t5v[4] || (x == t5v[4] && mi < t5i[4])) {
        int j = 4;
        while (j > 0 && (x > t5v[j - 1] || (x == t5v[j - 1] && mi < t5i[j - 1]))) {
          t5v[j] = t5v[j - 1]; t5i[j] = t5i[j - 1]; --j;
        }
        t5v[j] = x; t5i[j] = mi;
      }
    }
    sfifth[wave] = t5v[4];
  }
  __syncthreads();
  float fifth = sfifth[wave];
  int cntv = 0;
  for (int m = lane; m < NNODE; m += 32) cntv += (arow[m] == fifth);
  cntv += __shfl_xor(cntv, 16, 32);
  cntv += __shfl_xor(cntv, 8, 32);
  cntv += __shfl_xor(cntv, 4, 32);
  cntv += __shfl_xor(cntv, 2, 32);
  cntv += __shfl_xor(cntv, 1, 32);
  if (lane == 0) {
    int outi[3]; float outv[3];
    if (cntv <= 3) {
      for (int j = 0; j < 3; ++j) { outi[j] = t5i[j]; outv[j] = t5v[j]; }
    } else {
      int used[5] = {0, 0, 0, 0, 0};
      for (int pick = 0; pick < 3; ++pick) {
        int best = -1, bestd = 1 << 30;
        for (int j = 0; j < 5; ++j) {
          if (!used[j]) {
            int d = abs(t5i[j] - n);
            if (d < bestd) { bestd = d; best = j; }
          }
        }
        used[best] = 1;
        outi[pick] = t5i[best];
        outv[pick] = t5v[best];
      }
    }
    for (int j = 0; j < 3; ++j) {
      selid[row * 3 + j] = outi[j];
      selval[row * 3 + j] = outv[j];
    }
  }
}

// ---------------------------------------------------------------------------
// 10) Global min/max of selected weights (24576 values, one block)
// ---------------------------------------------------------------------------
__global__ __launch_bounds__(1024) void k_minmax(const float* __restrict__ w,
                                                 float* __restrict__ mm) {
  __shared__ float smn[1024], smx[1024];
  int t = threadIdx.x;
  float mn = 1e30f, mx = -1e30f;
  for (int i = t; i < NSEG * 3; i += 1024) {
    float x = w[i];
    mn = fminf(mn, x);
    mx = fmaxf(mx, x);
  }
  smn[t] = mn; smx[t] = mx;
  __syncthreads();
  for (int off = 512; off > 0; off >>= 1) {
    if (t < off) {
      smn[t] = fminf(smn[t], smn[t + off]);
      smx[t] = fmaxf(smx[t], smx[t + off]);
    }
    __syncthreads();
  }
  if (t == 0) { mm[0] = smn[0]; mm[1] = smx[0]; }
}

// ---------------------------------------------------------------------------
// 11) Finalize: edge list (ints) + normalized weights into d_out
// ---------------------------------------------------------------------------
__global__ void k_finalize(const int* __restrict__ selid,
                           const float* __restrict__ selval,
                           const float* __restrict__ mm,
                           int* __restrict__ out_i,
                           float* __restrict__ out_f) {
  int e = blockIdx.x * 256 + threadIdx.x;
  if (e >= NSEG * 3) return;
  int rn = e / 3;
  int r = rn >> 11;
  int n = rn & (NNODE - 1);
  out_i[e * 3 + 0] = n;
  out_i[e * 3 + 1] = selid[e];
  out_i[e * 3 + 2] = r;
  float mn = mm[0], mx = mm[1];
  out_f[e] = (selval[e] - mn) / (mx - mn + EPSF);
}

// ---------------------------------------------------------------------------
// Host launch
// ---------------------------------------------------------------------------
extern "C" void kernel_launch(void* const* d_in, const int* in_sizes, int n_in,
                              void* d_out, int out_size, void* d_ws,
                              size_t ws_size, hipStream_t stream) {
  const float* nf   = (const float*)d_in[0];
  const int*   el   = (const int*)d_in[1];
  const float* ewt  = (const float*)d_in[2];
  const float* Wlin = (const float*)d_in[3];
  const float* blin = (const float*)d_in[4];
  const float* Wq   = (const float*)d_in[5];
  const float* bq   = (const float*)d_in[6];
  const float* Wk   = (const float*)d_in[7];
  const float* bk   = (const float*)d_in[8];

  char* p = (char*)d_ws;
  float*     deg    = (float*)p;                 p += NSEG * 4;
  unsigned*  cnt    = (unsigned*)p;              p += NSEG * 4;
  unsigned*  offs   = (unsigned*)p;              p += NSEG * 4;
  unsigned*  cursor = (unsigned*)p;              p += NSEG * 4;
  unsigned*  perm   = (unsigned*)p;              p += (size_t)NEDGE * 4;
  _Float16*  Wlh    = (_Float16*)p;              p += (size_t)DOUT * DIN * 2;
  _Float16*  Wqh    = (_Float16*)p;              p += (size_t)DOUT * DIN * 2;
  _Float16*  Wkh    = (_Float16*)p;              p += (size_t)DOUT * DIN * 2;
  _Float16*  updh   = (_Float16*)p;              p += (size_t)NSEG * DIN * 2;
  _Float16*  feath  = (_Float16*)p;              p += (size_t)NSEG * DOUT * 2;
  _Float16*  Qh     = (_Float16*)p;              p += (size_t)NSEG * DOUT * 2;
  _Float16*  Kh     = (_Float16*)p;              p += (size_t)NSEG * DOUT * 2;
  float*     stats  = (float*)p;                 p += (size_t)NSEG * NHEAD * 2 * 4;
  float*     Am     = (float*)p;                 p += (size_t)NREL * NNODE * NNODE * 4;
  int*       selid  = (int*)p;                   p += NSEG * 3 * 4;
  float*     selval = (float*)p;                 p += NSEG * 3 * 4;
  float*     mm     = (float*)p;                 p += 256;

  hipMemsetAsync(deg, 0, NSEG * 4, stream);
  hipMemsetAsync(cnt, 0, NSEG * 4, stream);

  // CSR build
  k_edge_hist<<<NEDGE / 256, 256, 0, stream>>>(el, ewt, deg, cnt);
  k_scan<<<1, 1024, 0, stream>>>(cnt, offs, cursor);
  k_scatter<<<NEDGE / 256, 256, 0, stream>>>(el, cursor, perm);

  // Weight conversions (independent of CSR)
  k_cvt<<<(DOUT * DIN + 255) / 256, 256, 0, stream>>>(Wlin, Wlh, DOUT * DIN);
  k_cvt<<<(DOUT * DIN + 255) / 256, 256, 0, stream>>>(Wq, Wqh, DOUT * DIN);
  k_cvt<<<(DOUT * DIN + 255) / 256, 256, 0, stream>>>(Wk, Wkh, DOUT * DIN);

  // Segment gather-sum -> f16 update matrix
  k_gather<<<NSEG, 128, 0, stream>>>(nf, el, ewt, deg, offs, cnt, perm, updh);

  // feat = relu(update @ Wlin^T + b);  Q/K = feat @ W{q,k}^T + b
  k_gemm_h<<<4096, 128, 0, stream>>>(updh, Wlh, blin, feath, 1);
  k_gemm_h<<<4096, 128, 0, stream>>>(feath, Wqh, bq, Qh, 0);
  k_gemm_h<<<4096, 128, 0, stream>>>(feath, Wkh, bk, Kh, 0);

  // Two-pass fused softmax/head-mean attention
  k_attn_stats<<<1024, 128, 0, stream>>>(Qh, Kh, stats);
  k_attn_amean<<<8192, 256, 0, stream>>>(Qh, Kh, stats, Am);

  // Top-3 per row, normalize, emit outputs
  k_select<<<NSEG / 4, 128, 0, stream>>>(Am, selid, selval);
  k_minmax<<<1, 1024, 0, stream>>>(selval, mm);
  int* out_i = (int*)d_out;
  float* out_f = (float*)d_out + NSEG * 3 * 3;  // after 73728 ints
  k_finalize<<<(NSEG * 3 + 255) / 256, 256, 0, stream>>>(selid, selval, mm,
                                                         out_i, out_f);
  (void)in_sizes; (void)n_in; (void)out_size; (void)ws_size;
}
